// GaussianImage_Cholesky_11355893531379
// MI455X (gfx1250) — compile-verified, matrix operand order checked by reading, not run on hardware
//
#include <hip/hip_runtime.h>
#include <hip/hip_bf16.h>

typedef __attribute__((ext_vector_type(16))) _Float16     v16h;
typedef __attribute__((ext_vector_type(8)))  float        v8f;
typedef __attribute__((ext_vector_type(2)))  float        v2f;
typedef __attribute__((ext_vector_type(8)))  unsigned int v8u;

#define IMG_W 512
#define IMG_H 512
#define NG    2048

// -log2(e): folded into coefficients so the sigma-WMMA directly yields
// log2(alpha_raw) = -log2(e)*sigma + log2(opac), and alpha is a bare v_exp_f32.
#define NEG_LOG2E (-1.4426950408889634f)
#define LOG2E     (1.4426950408889634f)

__device__ __forceinline__ float fast_sigmoid(float v) {
    // 1 / (1 + exp(-v)) with raw v_exp_f32 / v_rcp_f32 (preprocessing only)
    return __builtin_amdgcn_rcpf(1.0f + __builtin_amdgcn_exp2f(-LOG2E * v));
}

// Per block: 256 threads = 8 wave32. Each wave renders a 16-pixel strip against
// all 2048 gaussians. sigma (plus folded log2-opacity) via v_wmma_f32_16x16x4_f32
// (rank-6 bilinear form, gaussian=M, pixel=N), whose D layout feeds the f16
// color-WMMA A layout with zero shuffles; RGB via v_wmma_f32_16x16x32_f16.
// Chunk loop unrolled 2x with independent accumulators to hide WMMA hazards.
//
// Note: the conic is positive-definite (det = l1^2*l3^2 > 0, s11 > 0), so
// sigma >= 0 holds mathematically for every pixel; the reference's
// where(sigma<0) clause is an fp-noise guard that we omit (keeping it against
// our differently-rounded basis evaluation would *add* deviation).
__global__ __launch_bounds__(256) void gsplat_wmma_kernel(
    const float* __restrict__ xyz,      // (1, N, 2)
    const float* __restrict__ chol,     // (1, N, 3)
    const float* __restrict__ opacity,  // (N, 1)
    const float* __restrict__ fdc,      // (N, 3)
    float* __restrict__ out)            // (3, H, W)
{
    __shared__ float    gcoef[NG * 6];   // scaled [a,b,c,d,e,f+log2(op)] (48 KB)
    __shared__ _Float16 colc[4 * NG];    // channel-major colors, ch3 = 0 (16 KB)

    const int tid = threadIdx.x;

    // ---- Preprocess all gaussians into LDS (8 per thread) ----
    for (int n = tid; n < NG; n += 256) {
        float mx = tanhf(xyz[2 * n + 0]);
        float my = tanhf(xyz[2 * n + 1]);
        float x  = 0.5f * (mx + 1.0f) * (float)IMG_W;
        float y  = 0.5f * (my + 1.0f) * (float)IMG_H;

        float l1 = chol[3 * n + 0] + 0.5f;
        float l2 = chol[3 * n + 1];
        float l3 = chol[3 * n + 2] + 0.5f;
        float s11 = l1 * l1;
        float s12 = l1 * l2;
        float s22 = l2 * l2 + l3 * l3;
        float det = s11 * s22 - s12 * s12;
        float inv = 1.0f / det;
        float a = 0.5f * s22 * inv;          // coeff of px^2
        float b = -s12 * inv;                // coeff of px*py
        float c = 0.5f * s11 * inv;          // coeff of py^2
        float d = -(2.0f * a * x + b * y);   // coeff of px
        float e = -(b * x + 2.0f * c * y);   // coeff of py
        float f = a * x * x + b * x * y + c * y * y;   // constant

        float op   = fast_sigmoid(opacity[n]);
        float l2op = __builtin_amdgcn_logf(op);        // raw v_log_f32 = log2(op)

        gcoef[n * 6 + 0] = NEG_LOG2E * a;
        gcoef[n * 6 + 1] = NEG_LOG2E * b;
        gcoef[n * 6 + 2] = NEG_LOG2E * c;
        gcoef[n * 6 + 3] = NEG_LOG2E * d;
        gcoef[n * 6 + 4] = NEG_LOG2E * e;
        gcoef[n * 6 + 5] = NEG_LOG2E * f + l2op;       // K5 row is constant 1

        colc[0 * NG + n] = (_Float16)fast_sigmoid(fdc[3 * n + 0]);
        colc[1 * NG + n] = (_Float16)fast_sigmoid(fdc[3 * n + 1]);
        colc[2 * NG + n] = (_Float16)fast_sigmoid(fdc[3 * n + 2]);
        colc[3 * NG + n] = (_Float16)0.0f;             // zero pad channel
    }
    __syncthreads();

    // ---- Per-wave strip setup ----
    const int lane   = tid & 31;
    const int wave   = tid >> 5;
    const int strip  = blockIdx.x * 8 + wave;   // 0 .. 16383
    const int pbase  = strip * 16;              // first pixel of strip
    const int m      = lane & 15;               // pixel row (also B/D column)
    const int p      = pbase + m;
    const float px   = (float)(p & (IMG_W - 1)) + 0.5f;
    const float py   = (float)(p >> 9) + 0.5f;
    const int laneHi = lane >> 4;               // 0 or 1
    const int koffA  = laneHi << 3;
    const int nCol   = m;

    // Loop-invariant pixel-basis B fragments for the sigma WMMAs.
    // f32 B 4x16 layout: VGPR0 = row K0 (lanes<16) / K2 (lanes>=16);
    //                    VGPR1 = row K1 / K3.
    const bool lo = (laneHi == 0);
    v2f Bp1, Bp2;
    Bp1.x = lo ? px * px : py * py;   // K0 = px^2   | K2 = py^2
    Bp1.y = lo ? px * py : px;        // K1 = px*py  | K3 = px
    Bp2.x = lo ? py : 0.0f;           // K4 = py     | K6 = 0 (pad row)
    Bp2.y = lo ? 1.0f : 0.0f;         // K5 = 1      | K7 = 0 (pad row)

    // Color-B column base: lanes with nCol>=3 read the zero-padded channel 3.
    const int           colBase = ((nCol < 3) ? nCol : 3) * NG;   // element index
    const unsigned int* colcW   = (const unsigned int*)colc;      // 2 f16 per word

    v8f acc0 = {};   // independent accumulators (even / odd chunks)
    v8f acc1 = {};

    auto do_chunk = [&](int kbase, v8f& acc) {
        // ---- log2(alpha_raw) via f32 WMMA: D(M=gaussian, N=pixel) ----
        const int g1 = kbase + m;        // gaussians 0..15 of chunk
        const int g2 = kbase + 16 + m;   // gaussians 16..31 of chunk

        v2f A1a = *(const v2f*)&gcoef[g1 * 6 + 2 * laneHi];   // (a,b) | (c,d)
        v2f A1b = *(const v2f*)&gcoef[g1 * 6 + 4];            // (e,f') | garbage*0
        v2f A2a = *(const v2f*)&gcoef[g2 * 6 + 2 * laneHi];
        v2f A2b = *(const v2f*)&gcoef[g2 * 6 + 4];

        v8f sig1 = {};
        sig1 = __builtin_amdgcn_wmma_f32_16x16x4_f32(false, A1a, false, Bp1, (short)0, sig1, false, false);
        sig1 = __builtin_amdgcn_wmma_f32_16x16x4_f32(false, A1b, false, Bp2, (short)0, sig1, false, false);
        v8f sig2 = {};
        sig2 = __builtin_amdgcn_wmma_f32_16x16x4_f32(false, A2a, false, Bp1, (short)0, sig2, false, false);
        sig2 = __builtin_amdgcn_wmma_f32_16x16x4_f32(false, A2b, false, Bp2, (short)0, sig2, false, false);
        // sig1[r] = log2(op) - log2(e)*sigma(gauss kbase+r+8*laneHi,    pixel m)
        // sig2[r] = log2(op) - log2(e)*sigma(gauss kbase+16+r+8*laneHi, pixel m)
        // => exactly the f16 A-matrix element order for the color WMMA.

        // ---- alpha per element: exp2 -> min -> threshold -> f16 ----
        v16h Af;
#pragma unroll
        for (int i = 0; i < 16; ++i) {
            float sc = (i < 8) ? sig1[i] : sig2[i - 8];      // log2(op*exp(-sigma))
            float al = __builtin_amdgcn_exp2f(sc);           // opac * exp(-sigma)
            al = fminf(al, 0.999f);
            al = (al < (1.0f / 255.0f)) ? 0.0f : al;
            Af[i] = (_Float16)al;
        }

        // ---- color B fragment (one aligned b32 word per VGPR, no masking) ----
        v8u Bu;
#pragma unroll
        for (int j = 0; j < 8; ++j) {
            int k0 = kbase + 2 * j + ((j >= 4) ? 8 : 0) + koffA;   // always even
            Bu[j] = colcW[(colBase + k0) >> 1];
        }
        v16h Bf = __builtin_bit_cast(v16h, Bu);

        // ---- RGB accumulation ----
        acc = __builtin_amdgcn_wmma_f32_16x16x32_f16(
            false, Af, false, Bf, (short)0, acc, false, false);
    };

    for (int ch = 0; ch < 64; ch += 2) {
        do_chunk(ch * 32, acc0);
        do_chunk(ch * 32 + 32, acc1);
    }

    // ---- Writeback: VGPR r -> (M = r + 8*laneHi, N = lane&15) ----
    if (nCol < 3) {
        const int rowOff = laneHi << 3;
#pragma unroll
        for (int r = 0; r < 8; ++r) {
            float v = acc0[r] + acc1[r];
            v = fminf(fmaxf(v, 0.0f), 1.0f);
            out[nCol * (IMG_H * IMG_W) + pbase + rowOff + r] = v;
        }
    }
}

extern "C" void kernel_launch(void* const* d_in, const int* in_sizes, int n_in,
                              void* d_out, int out_size, void* d_ws, size_t ws_size,
                              hipStream_t stream) {
    const float* xyz     = (const float*)d_in[0];  // (1, 2048, 2)
    const float* chol    = (const float*)d_in[1];  // (1, 2048, 3)
    const float* opacity = (const float*)d_in[2];  // (2048, 1)
    const float* fdc     = (const float*)d_in[3];  // (2048, 3)
    // d_in[4] = frame_index (T == 1, always 0)
    float* out = (float*)d_out;                    // (1, 3, 512, 512)

    dim3 grid((IMG_W * IMG_H) / (16 * 8));         // 2048 blocks
    dim3 block(256);                               // 8 wave32 per block
    gsplat_wmma_kernel<<<grid, block, 0, stream>>>(xyz, chol, opacity, fdc, out);
}
